// GeometricAttention_83373905150277
// MI455X (gfx1250) — matrix operand order repaired and checked by reading, not compiled, not run on hardware
//
#include <hip/hip_runtime.h>
#include <hip/hip_bf16.h>

#define D_MODEL   1024
#define NHEAD     16
#define HEAD_DIM  64
#define DIST_BINS 32
#define SEQ_B     2
#define SEQ_L     2048

typedef __bf16 bf16_t;
typedef __attribute__((ext_vector_type(16))) __bf16 v16bf;
typedef __attribute__((ext_vector_type(8)))  __bf16 v8bf;
typedef __attribute__((ext_vector_type(8)))  float  v8f;
typedef int vint4 __attribute__((vector_size(16)));

#ifndef __has_builtin
#define __has_builtin(x) 0
#endif
#if __has_builtin(__builtin_amdgcn_global_load_async_to_lds_b128) && \
    __has_builtin(__builtin_amdgcn_s_wait_asynccnt)
#define USE_ASYNC_LDS 1
#else
#define USE_ASYNC_LDS 0
#endif

#define GLOBAL_AS __attribute__((address_space(1)))
#define LDS_AS    __attribute__((address_space(3)))

// param types per hipcc diagnostic: vector_size(16) int pointer, AS1 (printed
// "__device__"); LDS side assumed symmetric in AS3. const dropped via integer
// round-trip; generic LDS address truncates to the 32-bit LDS offset.
static __device__ __forceinline__ GLOBAL_AS vint4* g128(const void* p) {
  return (GLOBAL_AS vint4*)(unsigned long long)(uintptr_t)p;
}
static __device__ __forceinline__ LDS_AS vint4* l128(void* p) {
  return (LDS_AS vint4*)(unsigned)(uintptr_t)p;
}

static __device__ __forceinline__ bf16_t f2bf(float x) { return (bf16_t)x; }

// A-matrix 16x32 bf16 fragment (ISA 7.12.2): lane holds row M=lane%16;
// elems 0..7 -> K = half*8 + j ; elems 8..15 -> K = 16 + half*8 + (j-8)
static __device__ __forceinline__ v16bf frag_a_lds(const bf16_t* row, int half) {
  v8bf lo = *(const v8bf*)(row + half * 8);
  v8bf hi = *(const v8bf*)(row + 16 + half * 8);
  v16bf r;
#pragma unroll
  for (int i = 0; i < 8; ++i) { r[i] = lo[i]; r[i + 8] = hi[i]; }
  return r;
}

// B-matrix 32x16 bf16 fragment: lane holds col N=lane%16; elems 0..7 -> K = half*8 + j ; elems 8..15 -> K = 16 + half*8 + (j-8)
static __device__ __forceinline__ v16bf frag_b_lds(const bf16_t* row, int half) {
  v8bf lo = *(const v8bf*)(row + half * 8);
  v8bf hi = *(const v8bf*)(row + 16 + half * 8);
  v16bf r;
#pragma unroll
  for (int i = 0; i < 8; ++i) { r[i] = lo[i]; r[i + 8] = hi[i]; }
  return r;
}

static __device__ __forceinline__ v8f v8f_zero() {
  v8f z = {0.f, 0.f, 0.f, 0.f, 0.f, 0.f, 0.f, 0.f};
  return z;
}

// ---------------------------------------------------------------------------
// Weight transpose + fp32 -> bf16:  W [K][N] row-major  ->  Wt [N][K]
// ---------------------------------------------------------------------------
__global__ __launch_bounds__(256) void wt_kernel(const float* __restrict__ W,
                                                 bf16_t* __restrict__ Wt) {
  __shared__ float t[32][33];
  const int nb = blockIdx.x * 32;
  const int kb = blockIdx.y * 32;
  const int tx = threadIdx.x & 31;
  const int ty = threadIdx.x >> 5;
#pragma unroll
  for (int i = 0; i < 32; i += 8)
    t[ty + i][tx] = W[(size_t)(kb + ty + i) * D_MODEL + nb + tx];   // t[k][n]
  __syncthreads();
#pragma unroll
  for (int i = 0; i < 32; i += 8)
    Wt[(size_t)(nb + ty + i) * D_MODEL + kb + tx] = f2bf(t[tx][ty + i]);
}

// ---------------------------------------------------------------------------
// Tiled GEMM:  Out[M x 1024] = A[M x 1024] * Wt^T + bias   (M = 4096)
// BM=128 BN=128 BK=64, 256 threads = 8 waves in 2(m) x 4(n); wave tile 64x32.
// ---------------------------------------------------------------------------
template <bool A_BF16, bool OUT_HEADS>
__global__ __launch_bounds__(256) void gemm_kernel(const void* __restrict__ Aptr,
                                                   const bf16_t* __restrict__ Wt,
                                                   const float* __restrict__ bias,
                                                   void* __restrict__ Out) {
  const int K = D_MODEL, N = D_MODEL;
  const int bm = blockIdx.y * 128;
  const int bn = blockIdx.x * 128;
  const int tid = threadIdx.x;
  const int lane = tid & 31, wid = tid >> 5;
  const int wm = (wid >> 2) * 64;
  const int wn = (wid & 3) * 32;
  const int half = lane >> 4, ln = lane & 15;

  __shared__ bf16_t lA[128 * 72];
  __shared__ bf16_t lB[128 * 72];

  v8f acc[4][2];
#pragma unroll
  for (int i = 0; i < 4; ++i)
#pragma unroll
    for (int j = 0; j < 2; ++j) acc[i][j] = v8f_zero();

  for (int k0 = 0; k0 < K; k0 += 64) {
    if (A_BF16) {
      const bf16_t* A = (const bf16_t*)Aptr;
#pragma unroll
      for (int it = 0; it < 4; ++it) {
        int idx = tid + it * 256;                 // 8-elt chunk id (8/row)
        int r = idx >> 3, c = (idx & 7) * 8;
#if USE_ASYNC_LDS
        __builtin_amdgcn_global_load_async_to_lds_b128(
            g128(A + (size_t)(bm + r) * K + k0 + c), l128(lA + r * 72 + c), 0, 0);
#else
        *(v8bf*)(lA + r * 72 + c) = *(const v8bf*)(A + (size_t)(bm + r) * K + k0 + c);
#endif
      }
    } else {
      const float* A = (const float*)Aptr;
#pragma unroll
      for (int it = 0; it < 8; ++it) {
        int idx = tid + it * 256;                 // float4 id (16/row)
        int r = idx >> 4, c = (idx & 15) * 4;
        float4 v = *(const float4*)(A + (size_t)(bm + r) * K + k0 + c);
        bf16_t* p = lA + r * 72 + c;
        p[0] = f2bf(v.x); p[1] = f2bf(v.y); p[2] = f2bf(v.z); p[3] = f2bf(v.w);
      }
    }
#pragma unroll
    for (int it = 0; it < 4; ++it) {
      int idx = tid + it * 256;
      int r = idx >> 3, c = (idx & 7) * 8;
#if USE_ASYNC_LDS
      __builtin_amdgcn_global_load_async_to_lds_b128(
          g128(Wt + (size_t)(bn + r) * K + k0 + c), l128(lB + r * 72 + c), 0, 0);
#else
      *(v8bf*)(lB + r * 72 + c) = *(const v8bf*)(Wt + (size_t)(bn + r) * K + k0 + c);
#endif
    }
#if USE_ASYNC_LDS
    __builtin_amdgcn_s_wait_asynccnt(0);
#endif
    __syncthreads();

    // prefetch next tiles into cache while WMMAs run
    if (k0 + 64 < K) {
      __builtin_prefetch((const char*)Wt +
                         ((size_t)(bn + (tid >> 1)) * K + k0 + 64 + (tid & 1) * 32) * 2, 0, 1);
      if (!A_BF16)
        __builtin_prefetch((const char*)Aptr +
                           ((size_t)(bm + (tid >> 1)) * K + k0 + 64 + (tid & 1) * 32) * 4, 0, 1);
      else
        __builtin_prefetch((const char*)Aptr +
                           ((size_t)(bm + (tid >> 1)) * K + k0 + 64 + (tid & 1) * 32) * 2, 0, 1);
    }

#pragma unroll
    for (int kk = 0; kk < 2; ++kk) {
      v16bf af[4], bfv[2];
#pragma unroll
      for (int mt = 0; mt < 4; ++mt)
        af[mt] = frag_a_lds(lA + (wm + mt * 16 + ln) * 72 + kk * 32, half);
#pragma unroll
      for (int nt = 0; nt < 2; ++nt)
        bfv[nt] = frag_b_lds(lB + (wn + nt * 16 + ln) * 72 + kk * 32, half);
#pragma unroll
      for (int mt = 0; mt < 4; ++mt)
#pragma unroll
        for (int nt = 0; nt < 2; ++nt)
          acc[mt][nt] = __builtin_amdgcn_wmma_f32_16x16x32_bf16(
              false, af[mt], false, bfv[nt], (short)0, acc[mt][nt], false, false);
    }
    __syncthreads();
  }

  // epilogue: C element (reg r, lane) -> M = r + 8*half, N = ln
#pragma unroll
  for (int mt = 0; mt < 4; ++mt) {
#pragma unroll
    for (int nt = 0; nt < 2; ++nt) {
      int n = bn + wn + nt * 16 + ln;
      float bv = bias[n];
#pragma unroll
      for (int r = 0; r < 8; ++r) {
        int m = bm + wm + mt * 16 + r + half * 8;
        float v = acc[mt][nt][r] + bv;
        if (OUT_HEADS) {
          int b = m >> 11, l = m & (SEQ_L - 1);
          int h = n >> 6, d = n & 63;
          ((bf16_t*)Out)[((((size_t)b * NHEAD + h) * SEQ_L + l) << 6) + d] = f2bf(v);
        } else {
          ((float*)Out)[(size_t)m * N + n] = v;
        }
      }
    }
  }
}

// ---------------------------------------------------------------------------
// Flash attention with geometric bias (exp2 domain). Block = 8 waves; each
// wave owns 16 q rows of a 128-row tile; stream K/V in 64-wide LDS tiles.
// ---------------------------------------------------------------------------
__global__ __launch_bounds__(256) void attn_kernel(const bf16_t* __restrict__ Q,
                                                   const bf16_t* __restrict__ Km,
                                                   const bf16_t* __restrict__ Vm,
                                                   const float* __restrict__ pos,
                                                   const float* __restrict__ dist_emb,
                                                   bf16_t* __restrict__ ctx) {
  const int b = blockIdx.z, h = blockIdx.y;
  const int q0 = blockIdx.x * 128;
  const int tid = threadIdx.x;
  const int lane = tid & 31, wid = tid >> 5;
  const int half = lane >> 4, ln = lane & 15;
  const float LOG2E = 1.44269504088896340736f;
  const float SCL = 0.125f * LOG2E;             // 1/sqrt(64) folded with log2(e)

  __shared__ bf16_t sQ[128 * 72];
  __shared__ bf16_t sK[64 * 72];
  __shared__ bf16_t sVt[64 * 72];               // transposed: [d][k]
  __shared__ bf16_t sP[128 * 72];
  __shared__ float  sQp[128 * 3];
  __shared__ float  sKp[64 * 3];
  __shared__ float  sEmb[DIST_BINS + 1];

  const size_t bh = ((size_t)b * NHEAD + h) * SEQ_L;

#pragma unroll
  for (int it = 0; it < 4; ++it) {
    int idx = tid + it * 256;
    int r = idx >> 3, c = (idx & 7) * 8;
#if USE_ASYNC_LDS
    __builtin_amdgcn_global_load_async_to_lds_b128(
        g128(Q + ((bh + q0 + r) << 6) + c), l128(sQ + r * 72 + c), 0, 0);
#else
    *(v8bf*)(sQ + r * 72 + c) = *(const v8bf*)(Q + ((bh + q0 + r) << 6) + c);
#endif
  }
  if (tid < 128) {
#pragma unroll
    for (int i = 0; i < 3; ++i)
      sQp[tid * 3 + i] = pos[((size_t)b * SEQ_L + q0 + tid) * 3 + i];
  }
  if (tid < DIST_BINS + 1) sEmb[tid] = dist_emb[tid * NHEAD + h] * LOG2E;

  float m_run[8], l_run[8];
  v8f oacc[4];
#pragma unroll
  for (int r = 0; r < 8; ++r) { m_run[r] = -3.0e38f; l_run[r] = 0.f; }
#pragma unroll
  for (int nt = 0; nt < 4; ++nt) oacc[nt] = v8f_zero();

  for (int k0 = 0; k0 < SEQ_L; k0 += 64) {
#pragma unroll
    for (int it = 0; it < 2; ++it) {
      int idx = tid + it * 256;
      int r = idx >> 3, c = (idx & 7) * 8;
#if USE_ASYNC_LDS
      __builtin_amdgcn_global_load_async_to_lds_b128(
          g128(Km + ((bh + k0 + r) << 6) + c), l128(sK + r * 72 + c), 0, 0);
#else
      *(v8bf*)(sK + r * 72 + c) = *(const v8bf*)(Km + ((bh + k0 + r) << 6) + c);
#endif
    }
#pragma unroll
    for (int it = 0; it < 2; ++it) {
      int idx = tid + it * 256;
      int r = idx >> 3, c = (idx & 7) * 8;
      v8bf v = *(const v8bf*)(Vm + ((bh + k0 + r) << 6) + c);
#pragma unroll
      for (int j = 0; j < 8; ++j) sVt[(c + j) * 72 + r] = v[j];
    }
    if (tid < 64) {
#pragma unroll
      for (int i = 0; i < 3; ++i)
        sKp[tid * 3 + i] = pos[((size_t)b * SEQ_L + k0 + tid) * 3 + i];
    }
#if USE_ASYNC_LDS
    __builtin_amdgcn_s_wait_asynccnt(0);
#endif
    __syncthreads();

    if (k0 + 64 < SEQ_L) {
      __builtin_prefetch((const char*)(Km + ((bh + k0 + 64) << 6)) + tid * 32, 0, 1);
      __builtin_prefetch((const char*)(Vm + ((bh + k0 + 64) << 6)) + tid * 32, 0, 1);
    }

    // S = Q * K^T  (wave rows wid*16..+15, 64 key columns = 4 subtiles)
    v8f s[4];
#pragma unroll
    for (int st = 0; st < 4; ++st) s[st] = v8f_zero();
#pragma unroll
    for (int kk = 0; kk < 2; ++kk) {
      v16bf aq = frag_a_lds(sQ + (wid * 16 + ln) * 72 + kk * 32, half);
#pragma unroll
      for (int st = 0; st < 4; ++st) {
        v16bf bk = frag_b_lds(sK + (st * 16 + ln) * 72 + kk * 32, half);
        s[st] = __builtin_amdgcn_wmma_f32_16x16x32_bf16(
            false, aq, false, bk, (short)0, s[st], false, false);
      }
    }

    // scale + geometric bias (log2 domain) + online softmax
    float sv[4][8], rmax[8];
#pragma unroll
    for (int r = 0; r < 8; ++r) rmax[r] = -3.0e38f;
#pragma unroll
    for (int st = 0; st < 4; ++st) {
      int krow = st * 16 + ln;
      float kx = sKp[krow * 3 + 0], ky = sKp[krow * 3 + 1], kz = sKp[krow * 3 + 2];
#pragma unroll
      for (int r = 0; r < 8; ++r) {
        int qrow = wid * 16 + r + half * 8;
        float dx = sQp[qrow * 3 + 0] - kx;
        float dy = sQp[qrow * 3 + 1] - ky;
        float dz = sQp[qrow * 3 + 2] - kz;
        float dist = __builtin_amdgcn_sqrtf(dx * dx + dy * dy + dz * dz);
        int bin = (int)(dist * ((float)DIST_BINS / 5.0f));
        bin = bin > DIST_BINS ? DIST_BINS : bin;
        float val = s[st][r] * SCL + sEmb[bin];
        sv[st][r] = val;
        rmax[r] = fmaxf(rmax[r], val);
      }
    }
#pragma unroll
    for (int msk = 1; msk < 16; msk <<= 1)
#pragma unroll
      for (int r = 0; r < 8; ++r)
        rmax[r] = fmaxf(rmax[r], __shfl_xor(rmax[r], msk, 32));

    float mnew[8], scal[8], rsum[8];
#pragma unroll
    for (int r = 0; r < 8; ++r) {
      mnew[r] = fmaxf(m_run[r], rmax[r]);
      scal[r] = __builtin_amdgcn_exp2f(m_run[r] - mnew[r]);
      rsum[r] = 0.f;
    }
#pragma unroll
    for (int st = 0; st < 4; ++st)
#pragma unroll
      for (int r = 0; r < 8; ++r) {
        float p = __builtin_amdgcn_exp2f(sv[st][r] - mnew[r]);
        sv[st][r] = p;
        rsum[r] += p;
      }
#pragma unroll
    for (int msk = 1; msk < 16; msk <<= 1)
#pragma unroll
      for (int r = 0; r < 8; ++r)
        rsum[r] += __shfl_xor(rsum[r], msk, 32);
#pragma unroll
    for (int r = 0; r < 8; ++r) {
      l_run[r] = l_run[r] * scal[r] + rsum[r];
      m_run[r] = mnew[r];
    }
#pragma unroll
    for (int nt = 0; nt < 4; ++nt)
#pragma unroll
      for (int r = 0; r < 8; ++r) oacc[nt][r] *= scal[r];

    // C-layout -> A-layout transpose of P via LDS
#pragma unroll
    for (int st = 0; st < 4; ++st)
#pragma unroll
      for (int r = 0; r < 8; ++r)
        sP[(wid * 16 + r + half * 8) * 72 + st * 16 + ln] = f2bf(sv[st][r]);
    __syncthreads();

    // O += P * V  (B operand from transposed V tile)
#pragma unroll
    for (int kk = 0; kk < 2; ++kk) {
      v16bf ap = frag_a_lds(sP + (wid * 16 + ln) * 72 + kk * 32, half);
#pragma unroll
      for (int nt = 0; nt < 4; ++nt) {
        v16bf bv = frag_b_lds(sVt + (nt * 16 + ln) * 72 + kk * 32, half);
        oacc[nt] = __builtin_amdgcn_wmma_f32_16x16x32_bf16(
            false, ap, false, bv, (short)0, oacc[nt], false, false);
      }
    }
    __syncthreads();
  }

  // normalize and write context in [B, L, H*hd] for the output GEMM
#pragma unroll
  for (int nt = 0; nt < 4; ++nt) {
    float rl[8];
#pragma unroll
    for (int r = 0; r < 8; ++r) rl[r] = __builtin_amdgcn_rcpf(l_run[r]);
#pragma unroll
    for (int r = 0; r < 8; ++r) {
      int qrow = q0 + wid * 16 + r + half * 8;
      int d = nt * 16 + ln;
      float v = oacc[nt][r] * rl[r];
      ctx[((size_t)b * SEQ_L + qrow) * D_MODEL + (h << 6) + d] = f2bf(v);
    }
  }
}

// ---------------------------------------------------------------------------
extern "C" void kernel_launch(void* const* d_in, const int* in_sizes, int n_in,
                              void* d_out, int out_size, void* d_ws, size_t ws_size,
                              hipStream_t stream) {
  const float* query = (const float*)d_in[0];
  const float* key   = (const float*)d_in[1];
  const float* value = (const float*)d_in[2];
  const float* posns = (const float*)d_in[3];
  const float* Wq    = (const float*)d_in[4];
  const float* bq    = (const float*)d_in[5];
  const float* Wk    = (const float*)d_in[6];
  const float* bk    = (const float*)d_in[7];
  const float* Wv    = (const float*)d_in[8];
  const float* bv    = (const float*)d_in[9];
  const float* Wo    = (const float*)d_in[10];
  const float* bo    = (const float*)d_in[11];
  const float* demb  = (const float*)d_in[12];

  bf16_t* ws = (bf16_t*)d_ws;
  const size_t WSZ = (size_t)D_MODEL * D_MODEL;               // 1M elems
  const size_t QSZ = (size_t)SEQ_B * SEQ_L * D_MODEL;         // 4M elems
  bf16_t* Wtq = ws;
  bf16_t* Wtk = ws + WSZ;
  bf16_t* Wtv = ws + 2 * WSZ;
  bf16_t* Wto = ws + 3 * WSZ;
  bf16_t* Qw  = ws + 4 * WSZ;
  bf16_t* Kw  = Qw + QSZ;
  bf16_t* Vw  = Kw + QSZ;
  bf16_t* Ctx = Vw + QSZ;

  dim3 wtg(32, 32);
  wt_kernel<<<wtg, 256, 0, stream>>>(Wq, Wtq);
  wt_kernel<<<wtg, 256, 0, stream>>>(Wk, Wtk);
  wt_kernel<<<wtg, 256, 0, stream>>>(Wv, Wtv);
  wt_kernel<<<wtg, 256, 0, stream>>>(Wo, Wto);

  dim3 gg(D_MODEL / 128, (SEQ_B * SEQ_L) / 128);   // 8 x 32
  gemm_kernel<false, true><<<gg, 256, 0, stream>>>(query, Wtq, bq, Qw);
  gemm_kernel<false, true><<<gg, 256, 0, stream>>>(key,   Wtk, bk, Kw);
  gemm_kernel<false, true><<<gg, 256, 0, stream>>>(value, Wtv, bv, Vw);

  dim3 ag(SEQ_L / 128, NHEAD, SEQ_B);              // 16 x 16 x 2
  attn_kernel<<<ag, 256, 0, stream>>>(Qw, Kw, Vw, posns, demb, Ctx);

  gemm_kernel<true, false><<<gg, 256, 0, stream>>>(Ctx, Wto, bo, (float*)d_out);
}
